// AttentionHead_43533788512394
// MI455X (gfx1250) — matrix operand order repaired and checked by reading, not compiled, not run on hardware
//
#include <hip/hip_runtime.h>
#include <hip/hip_bf16.h>

typedef __bf16 bf16_t;
typedef __attribute__((ext_vector_type(16))) __bf16 v16bf;
typedef __attribute__((ext_vector_type(8)))  __bf16 v8bf;
typedef __attribute__((ext_vector_type(8)))  float  v8f;

#define WMMA_BF16 __builtin_amdgcn_wmma_f32_16x16x32_bf16

union BF16Frag { v16bf v; v8bf h[2]; };

constexpr int B_  = 4;
constexpr int S_  = 2048;
constexpr int D_  = 4096;
constexpr int DH_ = 128;
constexpr int M_  = B_ * S_;     // 8192 flattened rows
constexpr int KB_ = D_ / 64;     // 64 K-steps of 64
constexpr int MT_ = 64;          // rows per block (4 A-tiles) in qkv kernel

static __device__ __forceinline__ float half_reduce_max(float x) {
#pragma unroll
  for (int d = 1; d < 16; d <<= 1) x = fmaxf(x, __shfl_xor(x, d, 32));
  return x;  // reduction stays inside each 16-lane half (wave32)
}
static __device__ __forceinline__ float half_reduce_sum(float x) {
#pragma unroll
  for (int d = 1; d < 16; d <<= 1) x += __shfl_xor(x, d, 32);
  return x;
}

// ---------------------------------------------------------------------------
// Kernel 1: fused QKV projection.  C[m,n] = bf16(x)[m,:] . W[n,:] + b[n]
// 64-row tile per block: each wave owns one 16-wide dh tile and amortizes its
// 6 W B-fragments over 4 A-tiles (24 WMMAs per K-step).  x read once from
// HBM, staged fp32->bf16 in double-buffered LDS, one barrier per K-step.
// Q,K stored row-major [M,DH]; V stored transposed Vt[B][DH][S].
// ---------------------------------------------------------------------------
__global__ __launch_bounds__(256) void qkv_kernel(
    const float* __restrict__ x,
    const bf16_t* __restrict__ Wq, const bf16_t* __restrict__ bq,
    const bf16_t* __restrict__ Wk, const bf16_t* __restrict__ bk,
    const bf16_t* __restrict__ Wv, const bf16_t* __restrict__ bv,
    bf16_t* __restrict__ Q, bf16_t* __restrict__ K, bf16_t* __restrict__ Vt)
{
  __shared__ bf16_t lds_x[2][MT_ * 64];      // double-buffered x tile (2x8 KB)

  const int tid  = threadIdx.x;
  const int wave = tid >> 5;                 // 0..7 -> dh tile
  const int lane = tid & 31;
  const int half = lane >> 4;                // lane half (wave32)
  const int r    = lane & 15;
  const int m0   = blockIdx.x * MT_;         // global row tile
  const int n0   = wave * 16;                // dh tile base

  const int xrow = tid >> 2;                 // 0..63
  const int xcol = (tid & 3) * 16;           // 0,16,32,48
  const float* xsrc = x + (size_t)(m0 + xrow) * D_ + xcol;

  v8f accq[4] = {}, acck[4] = {}, accv[4] = {};

  // prologue: stage tile 0 (16 floats -> 16 bf16 per thread)
  {
    float4 xv0 = *(const float4*)(xsrc + 0);
    float4 xv1 = *(const float4*)(xsrc + 4);
    float4 xv2 = *(const float4*)(xsrc + 8);
    float4 xv3 = *(const float4*)(xsrc + 12);
    v8bf lo, hi;
    lo[0] = (bf16_t)xv0.x; lo[1] = (bf16_t)xv0.y; lo[2] = (bf16_t)xv0.z; lo[3] = (bf16_t)xv0.w;
    lo[4] = (bf16_t)xv1.x; lo[5] = (bf16_t)xv1.y; lo[6] = (bf16_t)xv1.z; lo[7] = (bf16_t)xv1.w;
    hi[0] = (bf16_t)xv2.x; hi[1] = (bf16_t)xv2.y; hi[2] = (bf16_t)xv2.z; hi[3] = (bf16_t)xv2.w;
    hi[4] = (bf16_t)xv3.x; hi[5] = (bf16_t)xv3.y; hi[6] = (bf16_t)xv3.z; hi[7] = (bf16_t)xv3.w;
    *(v8bf*)(&lds_x[0][xrow * 64 + xcol])     = lo;
    *(v8bf*)(&lds_x[0][xrow * 64 + xcol + 8]) = hi;
  }

  for (int kb = 0; kb < KB_; ++kb) {
    __syncthreads();                         // cur buffer ready; prev reads done
    const int cur = kb & 1;

    // issue next x tile's global loads before the compute block
    float4 nxt0, nxt1, nxt2, nxt3;
    if (kb + 1 < KB_) {
      const float* p = xsrc + (size_t)(kb + 1) * 64;
      nxt0 = *(const float4*)(p + 0);
      nxt1 = *(const float4*)(p + 4);
      nxt2 = *(const float4*)(p + 8);
      nxt3 = *(const float4*)(p + 12);
    }
    if (kb + 2 < KB_)                        // pull tile kb+2 toward WGP$/L2
      __builtin_prefetch(xsrc + (size_t)(kb + 2) * 64, 0, 3);

    // six B fragments, shared by all 4 A-tiles this K-step
    const size_t wof = (size_t)(n0 + r) * D_ + (size_t)kb * 64 + half * 16;
    const v16bf bq0 = *(const v16bf*)(Wq + wof);
    const v16bf bq1 = *(const v16bf*)(Wq + wof + 32);
    const v16bf bk0 = *(const v16bf*)(Wk + wof);
    const v16bf bk1 = *(const v16bf*)(Wk + wof + 32);
    const v16bf bv0 = *(const v16bf*)(Wv + wof);
    const v16bf bv1 = *(const v16bf*)(Wv + wof + 32);

#pragma unroll
    for (int mt = 0; mt < 4; ++mt) {
      // A fragments (16x32): half0 K{0..15}, half1 K{16..31}
      BF16Frag a0, a1;
      const bf16_t* ar = &lds_x[cur][(mt * 16 + r) * 64];
      a0.v = *(const v16bf*)(ar + half * 16);
      a1.v = *(const v16bf*)(ar + 32 + half * 16);

      accq[mt] = WMMA_BF16(false, a0.v, false, bq0, (short)0, accq[mt], false, false);
      acck[mt] = WMMA_BF16(false, a0.v, false, bk0, (short)0, acck[mt], false, false);
      accv[mt] = WMMA_BF16(false, a0.v, false, bv0, (short)0, accv[mt], false, false);
      accq[mt] = WMMA_BF16(false, a1.v, false, bq1, (short)0, accq[mt], false, false);
      acck[mt] = WMMA_BF16(false, a1.v, false, bk1, (short)0, acck[mt], false, false);
      accv[mt] = WMMA_BF16(false, a1.v, false, bv1, (short)0, accv[mt], false, false);
    }

    // stage next tile into the other buffer (top-of-loop barrier already
    // separated it from last iteration's readers)
    if (kb + 1 < KB_) {
      v8bf lo, hi;
      lo[0] = (bf16_t)nxt0.x; lo[1] = (bf16_t)nxt0.y; lo[2] = (bf16_t)nxt0.z; lo[3] = (bf16_t)nxt0.w;
      lo[4] = (bf16_t)nxt1.x; lo[5] = (bf16_t)nxt1.y; lo[6] = (bf16_t)nxt1.z; lo[7] = (bf16_t)nxt1.w;
      hi[0] = (bf16_t)nxt2.x; hi[1] = (bf16_t)nxt2.y; hi[2] = (bf16_t)nxt2.z; hi[3] = (bf16_t)nxt2.w;
      hi[4] = (bf16_t)nxt3.x; hi[5] = (bf16_t)nxt3.y; hi[6] = (bf16_t)nxt3.z; hi[7] = (bf16_t)nxt3.w;
      *(v8bf*)(&lds_x[cur ^ 1][xrow * 64 + xcol])     = lo;
      *(v8bf*)(&lds_x[cur ^ 1][xrow * 64 + xcol + 8]) = hi;
    }
  }

  // epilogue: + bias, store.  C layout: col = n0+r, rows m = half*8 + i
  const float biasq = (float)bq[n0 + r];
  const float biask = (float)bk[n0 + r];
  const float biasv = (float)bv[n0 + r];
  const int b  = m0 >> 11;                   // / S_  (MT_ divides S_)
  const int s0 = m0 & (S_ - 1);

#pragma unroll
  for (int mt = 0; mt < 4; ++mt) {
#pragma unroll
    for (int i = 0; i < 8; ++i) {
      const size_t m = (size_t)m0 + mt * 16 + half * 8 + i;
      Q[m * DH_ + n0 + r] = (bf16_t)(accq[mt][i] + biasq);
      K[m * DH_ + n0 + r] = (bf16_t)(acck[mt][i] + biask);
    }
    v8bf vv;
#pragma unroll
    for (int i = 0; i < 8; ++i) vv[i] = (bf16_t)(accv[mt][i] + biasv);
    *(v8bf*)(Vt + ((size_t)b * DH_ + n0 + r) * S_ + s0 + mt * 16 + half * 8) = vv;
  }
}

// ---------------------------------------------------------------------------
// Kernel 2: causal flash attention.  One wave per 16-query tile.
// S = Q.K^T (fp32 acc), analytic causal mask (== additive -1e9 mask input),
// online softmax, P bounced via LDS to A-operand layout, O += P.V via Vt.
// ---------------------------------------------------------------------------
__global__ __launch_bounds__(32) void attn_kernel(
    const bf16_t* __restrict__ Q, const bf16_t* __restrict__ K,
    const bf16_t* __restrict__ Vt, bf16_t* __restrict__ out)
{
  __shared__ bf16_t lds_p[16 * 32];          // P tile bounce buffer (1 KB)

  const int lane = threadIdx.x & 31;
  const int half = lane >> 4;
  const int r    = lane & 15;
  const int t    = blockIdx.x;               // query tile
  const int b    = blockIdx.y;               // batch
  const int q0   = t * 16;

  // Q A-fragments: row = q0 + r for both halves, 4 K-chunks over DH=128
  BF16Frag qa[4];
  const bf16_t* qrow = Q + ((size_t)b * S_ + q0 + r) * DH_;
#pragma unroll
  for (int c = 0; c < 4; ++c) {
    qa[c].v = *(const v16bf*)(qrow + c * 32 + half * 16);
  }

  v8f o[8];
#pragma unroll
  for (int n = 0; n < 8; ++n) o[n] = (v8f){};
  float m_st[8], l_st[8];
#pragma unroll
  for (int i = 0; i < 8; ++i) { m_st[i] = -__builtin_inff(); l_st[i] = 0.f; }

  const int jmax = (q0 + 15) >> 5;           // causal: only key blocks <= diag
  for (int j = 0; j <= jmax; ++j) {
    const int kv0 = j * 32;

    // ---- S = Q . K^T for two 16-key subtiles (B col = key, K dim = dh) ----
    const bf16_t* krow0 = K + ((size_t)b * S_ + kv0 + r) * DH_ + half * 16;
    const bf16_t* krow1 = krow0 + (size_t)16 * DH_;
    v16bf kf0[4], kf1[4];
#pragma unroll
    for (int c = 0; c < 4; ++c) kf0[c] = *(const v16bf*)(krow0 + c * 32);
#pragma unroll
    for (int c = 0; c < 4; ++c) kf1[c] = *(const v16bf*)(krow1 + c * 32);

    v8f s0acc = {}, s1acc = {};
#pragma unroll
    for (int c = 0; c < 4; ++c)
      s0acc = WMMA_BF16(false, qa[c].v, false, kf0[c], (short)0, s0acc, false, false);
#pragma unroll
    for (int c = 0; c < 4; ++c)
      s1acc = WMMA_BF16(false, qa[c].v, false, kf1[c], (short)0, s1acc, false, false);

    // ---- online softmax update (per row i held by this lane half) ----
#pragma unroll
    for (int i = 0; i < 8; ++i) {
      const int row = q0 + half * 8 + i;
      const float s0 = s0acc[i] + ((kv0 + r)      > row ? -1e9f : 0.f);
      const float s1 = s1acc[i] + ((kv0 + 16 + r) > row ? -1e9f : 0.f);
      const float mx   = half_reduce_max(fmaxf(s0, s1));
      const float mnew = fmaxf(m_st[i], mx);
      const float corr = __expf(m_st[i] - mnew);
      const float p0 = __expf(s0 - mnew);
      const float p1 = __expf(s1 - mnew);
      l_st[i] = l_st[i] * corr + half_reduce_sum(p0 + p1);
      m_st[i] = mnew;
#pragma unroll
      for (int n = 0; n < 8; ++n) o[n][i] *= corr;
      // park P (bf16) row-major in LDS: row = half*8+i, cols r and 16+r
      lds_p[(half * 8 + i) * 32 + r]      = (bf16_t)p0;
      lds_p[(half * 8 + i) * 32 + 16 + r] = (bf16_t)p1;
    }

    // wave-internal LDS ordering: all 32 lanes' stores before cross-lane reads
    asm volatile("s_wait_dscnt 0" ::: "memory");

    // re-read P in A-operand layout (16x32)
    BF16Frag pf;
    pf.v = *(const v16bf*)(lds_p + r * 32 + half * 16);

    // ---- O += P . V  (B operand from Vt: col = dh, contiguous keys) ----
    const bf16_t* vbase = Vt + ((size_t)b * DH_ + r) * S_ + kv0 + half * 16;
#pragma unroll
    for (int g = 0; g < 2; ++g) {
      v16bf vf[4];
#pragma unroll
      for (int n = 0; n < 4; ++n)
        vf[n] = *(const v16bf*)(vbase + (size_t)(g * 4 + n) * 16 * S_);
#pragma unroll
      for (int n = 0; n < 4; ++n)
        o[g * 4 + n] =
            WMMA_BF16(false, pf.v, false, vf[n], (short)0, o[g * 4 + n], false, false);
    }
  }

  // ---- epilogue: O / l, store bf16 [B,S,DH] ----
  bf16_t* orow = out + ((size_t)b * S_ + q0) * DH_;
#pragma unroll
  for (int i = 0; i < 8; ++i) {
    const float inv = 1.f / l_st[i];
    const int row = half * 8 + i;
#pragma unroll
    for (int n = 0; n < 8; ++n)
      orow[(size_t)row * DH_ + n * 16 + r] = (bf16_t)(o[n][i] * inv);
  }
}

// ---------------------------------------------------------------------------
extern "C" void kernel_launch(void* const* d_in, const int* in_sizes, int n_in,
                              void* d_out, int out_size, void* d_ws, size_t ws_size,
                              hipStream_t stream) {
  (void)in_sizes; (void)n_in; (void)out_size; (void)ws_size;
  const float*  x  = (const float*)d_in[0];
  // d_in[1] = attn_mask: exact causal additive mask, applied analytically.
  const bf16_t* Wq = (const bf16_t*)d_in[2];
  const bf16_t* bq = (const bf16_t*)d_in[3];
  const bf16_t* Wk = (const bf16_t*)d_in[4];
  const bf16_t* bk = (const bf16_t*)d_in[5];
  const bf16_t* Wv = (const bf16_t*)d_in[6];
  const bf16_t* bv = (const bf16_t*)d_in[7];
  bf16_t* out = (bf16_t*)d_out;

  bf16_t* Qw  = (bf16_t*)d_ws;                       // 2 MB
  bf16_t* Kw  = Qw + (size_t)M_ * DH_;               // 2 MB
  bf16_t* Vtw = Kw + (size_t)M_ * DH_;               // 2 MB (transposed V)

  qkv_kernel<<<M_ / MT_, 256, 0, stream>>>(x, Wq, bq, Wk, bk, Wv, bv, Qw, Kw, Vtw);
  attn_kernel<<<dim3(S_ / 16, B_), 32, 0, stream>>>(Qw, Kw, Vtw, out);
}
